// SoftTree_21079699488961
// MI455X (gfx1250) — compile-verified, matrix operand order checked by reading, not run on hardware
//
#include <hip/hip_runtime.h>
#include <hip/hip_bf16.h>

typedef __attribute__((ext_vector_type(16))) _Float16 v16h;
typedef __attribute__((ext_vector_type(8)))  float    v8f;

#define BATCH    16384
#define IN_DIM   1024
#define NODES    1023
#define LEAVES   1024
#define OUT_DIM  128
#define DEPTH    10
#define MT       16            // batch rows per block
#define XT_STRIDE 1024         // halves per LDS row (16*1024*2 = 32 KB)

// ---- one-time precision conversion kernels (run every launch; deterministic) ----

// W (1023 x 1024 f32, row-major) -> Wh (1024 x 1024 f16, row 1023 zero-padded)
__global__ __launch_bounds__(256) void cvt_weights(const float* __restrict__ W,
                                                   _Float16* __restrict__ Wh) {
  int i = blockIdx.x * 256 + threadIdx.x;     // covers 1024*1024
  int row = i >> 10;
  int col = i & 1023;
  float v = (row < NODES) ? W[(size_t)row * IN_DIM + col] : 0.0f;
  Wh[i] = (_Float16)v;
}

// leaf_values (1024 x 128 f32) -> lvT (128 x 1024 f16, transposed so B-fragment
// lane loads are contiguous 32-byte reads)
__global__ __launch_bounds__(256) void cvt_leaves(const float* __restrict__ L,
                                                  _Float16* __restrict__ LT) {
  int i = blockIdx.x * 256 + threadIdx.x;     // covers 128*1024
  int c = i >> 10;                            // output column
  int l = i & 1023;                           // leaf index
  LT[i] = (_Float16)L[(size_t)l * OUT_DIM + c];
}

// ---- fused soft-tree kernel: decisions GEMM -> sigmoid -> routing -> leaf GEMM ----
__global__ __launch_bounds__(256) void soft_tree_fused(
    const float*    __restrict__ x,     // (16384, 1024) f32
    const _Float16* __restrict__ Wh,    // (1024, 1024) f16 (padded)
    const float*    __restrict__ bias,  // (1023,) f32
    const _Float16* __restrict__ lvT,   // (128, 1024) f16 transposed
    const float*    __restrict__ temp,  // scalar
    float*          __restrict__ out)   // (16384, 128) f32
{
  __shared__ __align__(32) _Float16 xt[MT * XT_STRIDE];   // x tile, later leaf probs
  __shared__ __align__(32) _Float16 dec[LEAVES * MT];     // decisions [node][row]

  const int tid  = threadIdx.x;
  const int lane = tid & 31;
  const int wave = tid >> 5;                 // 8 waves
  const int nsub = lane & 15;                // N index of 16x16 tile (also A row)
  const int ksub = (lane >> 4) * 16;         // 16-half K slice select
  const long row0 = (long)blockIdx.x * MT;
  const float invT = 1.0f / temp[0];

  // ---- stage x tile: 16 rows x 1024 f32 -> f16 LDS, NaN -> 0 ----
  {
    const float4* xg = (const float4*)(x + row0 * IN_DIM);
    for (int i = tid; i < MT * IN_DIM / 4; i += 256) {   // 4096 float4
      int r = i >> 8;                                    // IN_DIM/4 = 256
      int c = i & 255;
      float4 v = xg[(size_t)r * 256 + c];
      float a0 = (v.x != v.x) ? 0.0f : v.x;
      float a1 = (v.y != v.y) ? 0.0f : v.y;
      float a2 = (v.z != v.z) ? 0.0f : v.z;
      float a3 = (v.w != v.w) ? 0.0f : v.w;
      int b = r * XT_STRIDE + c * 4;
      xt[b + 0] = (_Float16)a0;
      xt[b + 1] = (_Float16)a1;
      xt[b + 2] = (_Float16)a2;
      xt[b + 3] = (_Float16)a3;
    }
  }
  __syncthreads();

  // ---- phase 1: decisions = sigmoid((x @ W^T + b) / T) via WMMA f16 ----
  // wave w owns node tiles {w, w+8, ..., w+56}; 32 k-steps of 16x16x32 each.
#pragma unroll 1
  for (int nt = wave; nt < 64; nt += 8) {
    const int node = nt * 16 + nsub;
    const _Float16* wrow = Wh + (size_t)node * IN_DIM + ksub;
    v8f acc = {};
#pragma unroll 1
    for (int k0 = 0; k0 < IN_DIM; k0 += 128) {
      // pull next node tile's W stream (L2-resident) toward the WGP; speculative,
      // unconditional -> no exec-mask branch in the hot loop.
      __builtin_prefetch(wrow + (size_t)128 * IN_DIM + k0, 0, 3);
#pragma unroll
      for (int kk = 0; kk < 128; kk += 32) {
        v16h a = *(const v16h*)(xt + nsub * XT_STRIDE + k0 + kk + ksub);
        v16h b = *(const v16h*)(wrow + k0 + kk);
        acc = __builtin_amdgcn_wmma_f32_16x16x32_f16(
            false, a, false, b, (short)0, acc, false, false);
      }
    }
    const float bb = (node < NODES) ? bias[node] : 0.0f;
#pragma unroll
    for (int v = 0; v < 8; ++v) {
      int m = v + ((lane >> 4) << 3);                    // batch row within tile
      float lg = (acc[v] + bb) * invT;
      float d  = 1.0f / (1.0f + __expf(-lg));
      dec[node * MT + m] = (_Float16)d;
    }
  }
  __syncthreads();

  // ---- phase 2: routing product -> leaf_probs (overwrites x tile) ----
  // thread handles row = tid&15, 64 leaves starting at (tid>>4)*64.
  {
    const int r = tid & 15;
    const int jb = (tid >> 4) * 64;
    for (int j = jb; j < jb + 64; ++j) {
      float p = 1.0f;
#pragma unroll
      for (int L = 0; L < DEPTH; ++L) {
        int mask = (1 << L) - 1;
        int node = mask + (j & mask);
        float d = (float)dec[node * MT + r];
        p *= ((j >> L) & 1) ? (1.0f - d) : d;
      }
      xt[r * XT_STRIDE + j] = (_Float16)p;
    }
  }
  __syncthreads();

  // ---- phase 3: out = leaf_probs @ leaf_values via WMMA f16 ----
  // wave w owns output columns [16w, 16w+16); 32 k-steps over 1024 leaves.
  {
    const int colb = wave * 16;
    const _Float16* lrow = lvT + (size_t)(colb + nsub) * LEAVES + ksub;
    v8f acc = {};
#pragma unroll 1
    for (int k0 = 0; k0 < LEAVES; k0 += 128) {
#pragma unroll
      for (int kk = 0; kk < 128; kk += 32) {
        v16h a = *(const v16h*)(xt + nsub * XT_STRIDE + k0 + kk + ksub);
        v16h b = *(const v16h*)(lrow + k0 + kk);
        acc = __builtin_amdgcn_wmma_f32_16x16x32_f16(
            false, a, false, b, (short)0, acc, false, false);
      }
    }
#pragma unroll
    for (int v = 0; v < 8; ++v) {
      int m = v + ((lane >> 4) << 3);
      out[(row0 + m) * OUT_DIM + colb + nsub] = acc[v];
    }
  }
}

extern "C" void kernel_launch(void* const* d_in, const int* in_sizes, int n_in,
                              void* d_out, int out_size, void* d_ws, size_t ws_size,
                              hipStream_t stream) {
  const float* x    = (const float*)d_in[0];   // (16384,1024)
  const float* W    = (const float*)d_in[1];   // (1023,1024)
  const float* b    = (const float*)d_in[2];   // (1023,)
  const float* lv   = (const float*)d_in[3];   // (1024,128)
  const float* temp = (const float*)d_in[4];   // scalar

  _Float16* Wh  = (_Float16*)d_ws;                       // 1024*1024 f16 = 2 MB
  _Float16* lvT = Wh + (size_t)1024 * 1024;              // 128*1024 f16 = 256 KB

  cvt_weights<<<(1024 * 1024) / 256, 256, 0, stream>>>(W, Wh);
  cvt_leaves <<<(128 * 1024) / 256,  256, 0, stream>>>(lv, lvT);
  soft_tree_fused<<<BATCH / MT, 256, 0, stream>>>(x, Wh, b, lvT, temp,
                                                  (float*)d_out);
}
